// AMIML_5102421148136
// MI455X (gfx1250) — compile-verified, matrix operand order checked by reading, not run on hardware
//
#include <hip/hip_runtime.h>
#include <hip/hip_bf16.h>
#include <math.h>

typedef __attribute__((ext_vector_type(16))) _Float16 v16h;
typedef __attribute__((ext_vector_type(8)))  _Float16 v8h;
typedef __attribute__((ext_vector_type(8)))  float    v8f;

#define BATCH 8
#define NPTS 100000
#define TILE_M 128
#define TILES_PER_B ((NPTS + TILE_M - 1) / TILE_M)   // 782
#define RSEL 100

// ---------------- workspace layout (bytes) ----------------
#define OFF_W1H   0u                    // 128*256 f16 = 65536
#define OFF_W2H   65536u                // 64*128 f16  = 16384
#define OFF_W3H   81920u                // 32*64 f16   = 4096
#define OFF_W4H   86016u                // 16*32 f16   = 1024 (padded cout 8->16)
#define OFF_SC1   87040u                // 128 f32
#define OFF_BI1   87552u
#define OFF_SC2   88064u                // 64 f32
#define OFF_BI2   88320u
#define OFF_SC3   88576u                // 32 f32
#define OFF_BI3   88704u
#define OFF_B4    88832u                // 16 f32
#define OFF_X4    89088u                // B*N*8 f32 = 25,600,000
#define OFF_S     25689088u             // B*N f32   = 3,200,000
#define OFF_TIDX  28889088u             // B*200 i32
#define OFF_TVAL  28895488u             // B*200 f32

// ---------------------------------------------------------------------------
// Prep: weights -> f16, fold BN into per-channel scale/bias
// ---------------------------------------------------------------------------
__global__ __launch_bounds__(256) void AMIML_prep(
    const float* __restrict__ c1w, const float* __restrict__ c1b,
    const float* __restrict__ g1, const float* __restrict__ be1,
    const float* __restrict__ m1, const float* __restrict__ v1,
    const float* __restrict__ c2w, const float* __restrict__ c2b,
    const float* __restrict__ g2, const float* __restrict__ be2,
    const float* __restrict__ m2, const float* __restrict__ v2,
    const float* __restrict__ c3w, const float* __restrict__ c3b,
    const float* __restrict__ g3, const float* __restrict__ be3,
    const float* __restrict__ m3, const float* __restrict__ v3,
    const float* __restrict__ c4w, const float* __restrict__ c4b,
    _Float16* __restrict__ w1h, _Float16* __restrict__ w2h,
    _Float16* __restrict__ w3h, _Float16* __restrict__ w4h,
    float* __restrict__ sc1, float* __restrict__ bi1,
    float* __restrict__ sc2, float* __restrict__ bi2,
    float* __restrict__ sc3, float* __restrict__ bi3,
    float* __restrict__ b4)
{
    int i = blockIdx.x * blockDim.x + threadIdx.x;
    if (i < 32768) { w1h[i] = (_Float16)c1w[i]; return; } i -= 32768;
    if (i < 8192)  { w2h[i] = (_Float16)c2w[i]; return; } i -= 8192;
    if (i < 2048)  { w3h[i] = (_Float16)c3w[i]; return; } i -= 2048;
    if (i < 512)   { int o = i >> 5, k = i & 31;
                     w4h[i] = (o < 8) ? (_Float16)c4w[o * 32 + k] : (_Float16)0.f; return; }
    i -= 512;
    if (i < 128) { float s = g1[i] * rsqrtf(v1[i] + 1e-5f);
                   sc1[i] = s; bi1[i] = (c1b[i] - m1[i]) * s + be1[i]; return; }
    i -= 128;
    if (i < 64)  { float s = g2[i] * rsqrtf(v2[i] + 1e-5f);
                   sc2[i] = s; bi2[i] = (c2b[i] - m2[i]) * s + be2[i]; return; }
    i -= 64;
    if (i < 32)  { float s = g3[i] * rsqrtf(v3[i] + 1e-5f);
                   sc3[i] = s; bi3[i] = (c3b[i] - m3[i]) * s + be3[i]; return; }
    i -= 32;
    if (i < 16)  { b4[i] = (i < 8) ? c4b[i] : 0.f; }
}

// ---------------------------------------------------------------------------
// WMMA fragment helpers (16-bit A/B layout: lanes 0-15 khalf=0, 16-31 khalf=8;
// elements 0..7 -> K = kc+khalf..+7, elements 8..15 -> K = kc+16+khalf..+7)
// ---------------------------------------------------------------------------
__device__ __forceinline__ v8f wmma16(v16h a, v16h b, v8f c) {
    return __builtin_amdgcn_wmma_f32_16x16x32_f16(
        false, a, false, b, (short)0, c, false, false);
}

__device__ __forceinline__ v16h load_frag_h(const _Float16* __restrict__ rowp,
                                            int kc, int khalf) {
    const v8h* p0 = (const v8h*)(rowp + kc + khalf);
    const v8h* p1 = (const v8h*)(rowp + kc + 16 + khalf);
    v8h lo = *p0, hi = *p1;
    v16h r;
#pragma unroll
    for (int e = 0; e < 8; ++e) { r[e] = lo[e]; r[8 + e] = hi[e]; }
    return r;
}

__device__ __forceinline__ v16h load_a_f32(const float* __restrict__ row,
                                           int kc, int khalf, bool valid) {
    v16h a;
    if (valid) {
        const float4* p0 = (const float4*)(row + kc + khalf);
        const float4* p1 = (const float4*)(row + kc + 16 + khalf);
        float4 a0 = p0[0], a1 = p0[1], a2 = p1[0], a3 = p1[1];
        a[0] = (_Float16)a0.x;  a[1] = (_Float16)a0.y;
        a[2] = (_Float16)a0.z;  a[3] = (_Float16)a0.w;
        a[4] = (_Float16)a1.x;  a[5] = (_Float16)a1.y;
        a[6] = (_Float16)a1.z;  a[7] = (_Float16)a1.w;
        a[8] = (_Float16)a2.x;  a[9] = (_Float16)a2.y;
        a[10] = (_Float16)a2.z; a[11] = (_Float16)a2.w;
        a[12] = (_Float16)a3.x; a[13] = (_Float16)a3.y;
        a[14] = (_Float16)a3.z; a[15] = (_Float16)a3.w;
    } else {
#pragma unroll
        for (int e = 0; e < 16; ++e) a[e] = (_Float16)0.f;
    }
    return a;
}

// ---------------------------------------------------------------------------
// Fused point-MLP: x[B,N,256] -> x4[B,N,8], s[B,N]  (WMMA f16 -> f32 accum)
// ---------------------------------------------------------------------------
__global__ __launch_bounds__(256) void AMIML_mlp(
    const float* __restrict__ x,
    const _Float16* __restrict__ w1h, const _Float16* __restrict__ w2h,
    const _Float16* __restrict__ w3h, const _Float16* __restrict__ w4h,
    const float* __restrict__ sc1, const float* __restrict__ bi1,
    const float* __restrict__ sc2, const float* __restrict__ bi2,
    const float* __restrict__ sc3, const float* __restrict__ bi3,
    const float* __restrict__ b4,
    const float* __restrict__ w5, const float* __restrict__ b5,
    float* __restrict__ x4g, float* __restrict__ sg)
{
    __shared__ _Float16 hA[8][16 * 128];   // per-wave ping (layer1 out, layer3 out)
    __shared__ _Float16 hB[8][16 * 64];    // per-wave pong (layer2 out)
    __shared__ float    x4s[8][16 * 8];

    const int tile = blockIdx.x;
    const int b    = tile / TILES_PER_B;
    const int n0   = (tile % TILES_PER_B) * TILE_M;
    const int wave = threadIdx.x >> 5;
    const int lane = threadIdx.x & 31;
    const int l15  = lane & 15;
    const int hi   = lane >> 4;
    const int khalf = hi * 8;

    const int rowA = n0 + wave * 16 + l15;
    const bool avalid = rowA < NPTS;
    const float* xrow = x + ((size_t)b * NPTS + rowA) * 256;
    if (avalid) __builtin_prefetch(xrow, 0, 1);

    _Float16* hAw = &hA[wave][0];
    _Float16* hBw = &hB[wave][0];
    float*    x4w = &x4s[wave][0];

    const v8f vzero = {0.f, 0.f, 0.f, 0.f, 0.f, 0.f, 0.f, 0.f};

    // ---- Layer 1: 256 -> 128 (8 K-chunks x 8 N-tiles)
    v8f acc1[8];
#pragma unroll
    for (int nt = 0; nt < 8; ++nt) acc1[nt] = vzero;
#pragma unroll
    for (int kt = 0; kt < 8; ++kt) {
        v16h a = load_a_f32(xrow, kt * 32, khalf, avalid);
#pragma unroll
        for (int nt = 0; nt < 8; ++nt) {
            v16h bf = load_frag_h(w1h + (size_t)(nt * 16 + l15) * 256, kt * 32, khalf);
            acc1[nt] = wmma16(a, bf, acc1[nt]);
        }
    }
#pragma unroll
    for (int nt = 0; nt < 8; ++nt) {
        int col = nt * 16 + l15;
        float sc = sc1[col], bi = bi1[col];
#pragma unroll
        for (int r = 0; r < 8; ++r) {
            float y = acc1[nt][r] * sc + bi;
            y = y > 0.f ? y : 0.f;
            hAw[(r + 8 * hi) * 128 + col] = (_Float16)y;
        }
    }

    // ---- Layer 2: 128 -> 64
    v8f acc2[4];
#pragma unroll
    for (int nt = 0; nt < 4; ++nt) acc2[nt] = vzero;
#pragma unroll
    for (int kt = 0; kt < 4; ++kt) {
        v16h a = load_frag_h(hAw + l15 * 128, kt * 32, khalf);
#pragma unroll
        for (int nt = 0; nt < 4; ++nt) {
            v16h bf = load_frag_h(w2h + (size_t)(nt * 16 + l15) * 128, kt * 32, khalf);
            acc2[nt] = wmma16(a, bf, acc2[nt]);
        }
    }
#pragma unroll
    for (int nt = 0; nt < 4; ++nt) {
        int col = nt * 16 + l15;
        float sc = sc2[col], bi = bi2[col];
#pragma unroll
        for (int r = 0; r < 8; ++r) {
            float y = acc2[nt][r] * sc + bi;
            y = y > 0.f ? y : 0.f;
            hBw[(r + 8 * hi) * 64 + col] = (_Float16)y;
        }
    }

    // ---- Layer 3: 64 -> 32
    v8f acc3[2];
#pragma unroll
    for (int nt = 0; nt < 2; ++nt) acc3[nt] = vzero;
#pragma unroll
    for (int kt = 0; kt < 2; ++kt) {
        v16h a = load_frag_h(hBw + l15 * 64, kt * 32, khalf);
#pragma unroll
        for (int nt = 0; nt < 2; ++nt) {
            v16h bf = load_frag_h(w3h + (size_t)(nt * 16 + l15) * 64, kt * 32, khalf);
            acc3[nt] = wmma16(a, bf, acc3[nt]);
        }
    }
#pragma unroll
    for (int nt = 0; nt < 2; ++nt) {
        int col = nt * 16 + l15;
        float sc = sc3[col], bi = bi3[col];
#pragma unroll
        for (int r = 0; r < 8; ++r) {
            float y = acc3[nt][r] * sc + bi;
            y = y > 0.f ? y : 0.f;
            hAw[(r + 8 * hi) * 32 + col] = (_Float16)y;   // reuse ping, stride 32
        }
    }

    // ---- Layer 4: 32 -> 8 (padded to 16 cols, no BN/ReLU)
    v8f acc4 = vzero;
    {
        v16h a = load_frag_h(hAw + l15 * 32, 0, khalf);
        v16h bf = load_frag_h(w4h + (size_t)l15 * 32, 0, khalf);
        acc4 = wmma16(a, bf, acc4);
    }
    if (l15 < 8) {
        float bias = b4[l15];
#pragma unroll
        for (int r = 0; r < 8; ++r)
            x4w[(r + 8 * hi) * 8 + l15] = acc4[r] + bias;
    }

    // ---- score head + spill x4/s
    if (lane < 16) {
        int n = n0 + wave * 16 + lane;
        if (n < NPTS) {
            const float* xr = x4w + lane * 8;
            float* og = x4g + ((size_t)b * NPTS + n) * 8;
            float sacc = b5[0];
#pragma unroll
            for (int d = 0; d < 8; ++d) { float v = xr[d]; og[d] = v; sacc += v * w5[d]; }
            sg[(size_t)b * NPTS + n] = sacc;
        }
    }
}

// ---------------------------------------------------------------------------
// Deterministic top-100 / bottom-100 per batch (iterative max extraction on
// packed (ordered_float, ~index) keys; tie-break = smaller index)
// ---------------------------------------------------------------------------
__device__ __forceinline__ unsigned ordf(float f) {
    unsigned u = __float_as_uint(f);
    return (u & 0x80000000u) ? ~u : (u | 0x80000000u);
}

__global__ __launch_bounds__(1024) void AMIML_select(
    const float* __restrict__ sg, int* __restrict__ tidx, float* __restrict__ tval)
{
    __shared__ unsigned long long red[1024];
    __shared__ unsigned long long prevS;
    const int b = blockIdx.x >> 1;
    const int which = blockIdx.x & 1;           // 0 = top(desc), 1 = bottom(asc)
    const float* s = sg + (size_t)b * NPTS;
    const int t = threadIdx.x;
    unsigned long long prev = ~0ull;

    for (int pass = 0; pass < RSEL; ++pass) {
        unsigned long long best = 0;
        for (int i = t; i < NPTS; i += 1024) {
            unsigned u = ordf(s[i]);
            unsigned hiw = which ? (0xFFFFFFFFu - u) : u;
            unsigned long long key =
                ((unsigned long long)hiw << 32) | (0xFFFFFFFFu - (unsigned)i);
            if (key < prev && key > best) best = key;
        }
        red[t] = best;
        __syncthreads();
        for (int strd = 512; strd > 0; strd >>= 1) {
            if (t < strd) { unsigned long long o = red[t + strd]; if (o > red[t]) red[t] = o; }
            __syncthreads();
        }
        if (t == 0) {
            unsigned long long bk = red[0];
            int idx = (int)(0xFFFFFFFFu - (unsigned)(bk & 0xFFFFFFFFu));
            int slot = b * 2 * RSEL + which * RSEL + pass;
            tidx[slot] = idx;
            tval[slot] = s[idx];
            prevS = bk;
        }
        __syncthreads();
        prev = prevS;
        __syncthreads();
    }
}

// ---------------------------------------------------------------------------
// Tiny attention epilogue: gather -> h2 -> q,k,v -> softmax attn -> pool -> softmax
// ---------------------------------------------------------------------------
__global__ __launch_bounds__(256) void AMIML_attn(
    const float* __restrict__ x4g, const int* __restrict__ tidx,
    const float* __restrict__ tval,
    const float* __restrict__ w6, const float* __restrict__ b6,
    const float* __restrict__ qw, const float* __restrict__ qb,
    const float* __restrict__ kw, const float* __restrict__ kb,
    const float* __restrict__ vw, const float* __restrict__ vb,
    float* __restrict__ out)
{
    __shared__ float h2S[200 * 8];
    __shared__ float kS[200 * 8];
    __shared__ float vS[200 * 8];
    __shared__ float pooled[8];
    const int b = blockIdx.x;
    const int t = threadIdx.x;

    if (t < 200) {
        int idx = tidx[b * 200 + t];
        float val = tval[b * 200 + t];
        const float* xr = x4g + ((size_t)b * NPTS + idx) * 8;
#pragma unroll
        for (int d = 0; d < 8; ++d)
            h2S[t * 8 + d] = 0.1f * (val * w6[d] + b6[d]) + xr[d];
    }
    __syncthreads();

    float qreg[8];
    if (t < 200) {
#pragma unroll
        for (int e = 0; e < 8; ++e) {
            float aq = qb[e], ak = kb[e], av = vb[e];
#pragma unroll
            for (int d = 0; d < 8; ++d) {
                float h = h2S[t * 8 + d];
                aq += h * qw[e * 8 + d];
                ak += h * kw[e * 8 + d];
                av += h * vw[e * 8 + d];
            }
            qreg[e] = aq; kS[t * 8 + e] = ak; vS[t * 8 + e] = av;
        }
    }
    __syncthreads();

    if (t < 200) {
        const float scale = 0.3535533905932738f;   // 1/sqrt(8)
        float mx = -3.0e38f;
        for (int m = 0; m < 200; ++m) {
            float d = 0.f;
#pragma unroll
            for (int e = 0; e < 8; ++e) d += qreg[e] * kS[m * 8 + e];
            d *= scale;
            mx = d > mx ? d : mx;
        }
        float sum = 0.f;
        float acc[8] = {0, 0, 0, 0, 0, 0, 0, 0};
        for (int m = 0; m < 200; ++m) {
            float d = 0.f;
#pragma unroll
            for (int e = 0; e < 8; ++e) d += qreg[e] * kS[m * 8 + e];
            float p = expf(d * scale - mx);
            sum += p;
#pragma unroll
            for (int e = 0; e < 8; ++e) acc[e] += p * vS[m * 8 + e];
        }
        float inv = 1.f / sum;
#pragma unroll
        for (int e = 0; e < 8; ++e) h2S[t * 8 + e] = acc[e] * inv;  // reuse as attn rows
    }
    __syncthreads();

    if (t < 8) {
        float a = 0.f;
        for (int j = 0; j < 200; ++j) a += h2S[j * 8 + t];
        pooled[t] = a * (1.f / 200.f);
    }
    __syncthreads();

    if (t == 0) {
        float mx = pooled[0];
#pragma unroll
        for (int d = 1; d < 8; ++d) mx = pooled[d] > mx ? pooled[d] : mx;
        float sum = 0.f;
        float ex[8];
#pragma unroll
        for (int d = 0; d < 8; ++d) { ex[d] = expf(pooled[d] - mx); sum += ex[d]; }
        float inv = 1.f / sum;
#pragma unroll
        for (int d = 0; d < 8; ++d) out[b * 8 + d] = ex[d] * inv;
    }
}

// ---------------------------------------------------------------------------
extern "C" void kernel_launch(void* const* d_in, const int* in_sizes, int n_in,
                              void* d_out, int out_size, void* d_ws, size_t ws_size,
                              hipStream_t stream) {
    (void)in_sizes; (void)n_in; (void)out_size; (void)ws_size;
    const float* x    = (const float*)d_in[0];
    const float* c1w  = (const float*)d_in[1];  const float* c1b = (const float*)d_in[2];
    const float* g1   = (const float*)d_in[3];  const float* be1 = (const float*)d_in[4];
    const float* m1   = (const float*)d_in[5];  const float* v1  = (const float*)d_in[6];
    const float* c2w  = (const float*)d_in[7];  const float* c2b = (const float*)d_in[8];
    const float* g2   = (const float*)d_in[9];  const float* be2 = (const float*)d_in[10];
    const float* m2   = (const float*)d_in[11]; const float* v2  = (const float*)d_in[12];
    const float* c3w  = (const float*)d_in[13]; const float* c3b = (const float*)d_in[14];
    const float* g3   = (const float*)d_in[15]; const float* be3 = (const float*)d_in[16];
    const float* m3   = (const float*)d_in[17]; const float* v3  = (const float*)d_in[18];
    const float* c4w  = (const float*)d_in[19]; const float* c4b = (const float*)d_in[20];
    const float* w5   = (const float*)d_in[25]; const float* b5  = (const float*)d_in[26];
    const float* w6   = (const float*)d_in[27]; const float* b6  = (const float*)d_in[28];
    const float* qw   = (const float*)d_in[29]; const float* qb  = (const float*)d_in[30];
    const float* kw   = (const float*)d_in[31]; const float* kb  = (const float*)d_in[32];
    const float* vw   = (const float*)d_in[33]; const float* vb  = (const float*)d_in[34];

    char* ws = (char*)d_ws;
    _Float16* w1h = (_Float16*)(ws + OFF_W1H);
    _Float16* w2h = (_Float16*)(ws + OFF_W2H);
    _Float16* w3h = (_Float16*)(ws + OFF_W3H);
    _Float16* w4h = (_Float16*)(ws + OFF_W4H);
    float* sc1 = (float*)(ws + OFF_SC1); float* bi1 = (float*)(ws + OFF_BI1);
    float* sc2 = (float*)(ws + OFF_SC2); float* bi2 = (float*)(ws + OFF_BI2);
    float* sc3 = (float*)(ws + OFF_SC3); float* bi3 = (float*)(ws + OFF_BI3);
    float* b4  = (float*)(ws + OFF_B4);
    float* x4g = (float*)(ws + OFF_X4);
    float* sg  = (float*)(ws + OFF_S);
    int*   tidx = (int*)(ws + OFF_TIDX);
    float* tval = (float*)(ws + OFF_TVAL);

    const int prep_total = 32768 + 8192 + 2048 + 512 + 128 + 64 + 32 + 16;
    AMIML_prep<<<(prep_total + 255) / 256, 256, 0, stream>>>(
        c1w, c1b, g1, be1, m1, v1, c2w, c2b, g2, be2, m2, v2,
        c3w, c3b, g3, be3, m3, v3, c4w, c4b,
        w1h, w2h, w3h, w4h, sc1, bi1, sc2, bi2, sc3, bi3, b4);

    AMIML_mlp<<<BATCH * TILES_PER_B, 256, 0, stream>>>(
        x, w1h, w2h, w3h, w4h, sc1, bi1, sc2, bi2, sc3, bi3, b4,
        w5, b5, x4g, sg);

    AMIML_select<<<BATCH * 2, 1024, 0, stream>>>(sg, tidx, tval);

    AMIML_attn<<<BATCH, 256, 0, stream>>>(
        x4g, tidx, tval, w6, b6, qw, qb, kw, kb, vw, vb, (float*)d_out);
}
